// Skip_NLM_59287728554356
// MI455X (gfx1250) — matrix operand order repaired and checked by reading, not compiled, not run on hardware
//
#include <hip/hip_runtime.h>
#include <hip/hip_bf16.h>

typedef __attribute__((ext_vector_type(16))) __bf16 v16bf;
typedef __attribute__((ext_vector_type(8)))  float  v8f;
typedef __attribute__((ext_vector_type(4)))  unsigned int v4u;

// ---------- bf16 helpers (raw ushort storage) ----------
__device__ __forceinline__ float bf2f(unsigned short h) {
    unsigned int u = ((unsigned int)h) << 16;
    return __builtin_bit_cast(float, u);
}
__device__ __forceinline__ unsigned short f2bf(float f) {
    unsigned int u = __builtin_bit_cast(unsigned int, f);
    u += 0x7FFFu + ((u >> 16) & 1u);   // round-to-nearest-even
    return (unsigned short)(u >> 16);
}

union FragAB { v4u q[2]; v16bf v; };

// =====================================================================
// Implicit-GEMM conv, NHWC bf16 in -> NHWC f32 out (+bias), cout = 128.
// Reflection padding (k-1)/2, stride 1 or 2. One wave computes a
// 16(M spatial) x 128(N cout) stripe with 8 bf16 WMMAs per K-chunk.
// B fragments are staged tap-by-tap through 32KB of LDS (4 K-chunks x
// 8 n-tiles x 1KB) so the 8 waves of the block share one global fetch.
// =====================================================================
__global__ void __launch_bounds__(256)
conv_wmma(const unsigned short* __restrict__ in,   // NHWC bf16
          const unsigned int*  __restrict__ wp,    // prepacked B fragments
          const float*         __restrict__ bias,  // [128]
          float*               __restrict__ out,   // (M,128) f32
          int N, int Hin, int Win, int Cin,
          int Hout, int Wout, int stride, int ksz)
{
    __shared__ v4u wlds[2048];                      // 32 KB: 32 fragments
    const int lane = threadIdx.x & 31;
    const int wave = threadIdx.x >> 5;
    const long Mtot = (long)N * Hout * Wout;
    const long mBase = ((long)blockIdx.x * 8 + wave) * 16;

    // This lane's A-matrix row (M = lane & 15 for both lane halves)
    long m = mBase + (lane & 15);
    long mm = (m < Mtot) ? m : (Mtot - 1);
    int n_  = (int)(mm / ((long)Hout * Wout));
    int rem = (int)(mm % ((long)Hout * Wout));
    int oh = rem / Wout, ow = rem % Wout;

    const int pad = (ksz - 1) >> 1;
    const int KC  = Cin >> 5;                       // K-chunks of 32 (4 or 8)
    const int khalf = (lane >= 16) ? 16 : 0;

    v8f acc[8];
#pragma unroll
    for (int t = 0; t < 8; ++t)
#pragma unroll
        for (int e = 0; e < 8; ++e) acc[t][e] = 0.0f;

    for (int kh = 0; kh < ksz; ++kh) {
        int ih = oh * stride + kh - pad;
        ih = (ih < 0) ? -ih : ih;
        ih = (ih >= Hin) ? (2 * Hin - 2 - ih) : ih;
        for (int kw = 0; kw < ksz; ++kw) {
            int iw = ow * stride + kw - pad;
            iw = (iw < 0) ? -iw : iw;
            iw = (iw >= Win) ? (2 * Win - 2 - iw) : iw;
            const unsigned short* src = in + (((long)n_ * Hin + ih) * Win + iw) * Cin;
            const int tap = kh * ksz + kw;
            for (int kcb = 0; kcb < KC; kcb += 4) {
                // ---- stage 4 K-chunks of B fragments into LDS ----
                __syncthreads();
                {
                    const v4u* gsrc = (const v4u*)(wp + ((long)(tap * KC + kcb) * 8) * 256);
#pragma unroll
                    for (int j = 0; j < 8; ++j)
                        wlds[threadIdx.x + j * 256] = gsrc[threadIdx.x + j * 256];
                }
                __syncthreads();
#pragma unroll
                for (int kc2 = 0; kc2 < 4; ++kc2) {
                    const int k0 = (kcb + kc2) * 32 + khalf;
                    FragAB A;
                    A.q[0] = *(const v4u*)(src + k0);        // channels k0..k0+7
                    A.q[1] = *(const v4u*)(src + k0 + 8);    // channels k0+8..k0+15
#pragma unroll
                    for (int nt = 0; nt < 8; ++nt) {
                        FragAB B;
                        const int fb = (kc2 * 8 + nt) * 64 + lane * 2;
                        B.q[0] = wlds[fb];
                        B.q[1] = wlds[fb + 1];
                        acc[nt] = __builtin_amdgcn_wmma_f32_16x16x32_bf16(
                            false, A.v, false, B.v, (short)0, acc[nt], false, false);
                    }
                }
            }
        }
    }

    // C/D layout: lane l, vgpr r -> M = r + (l<16?0:8), N = l & 15
    const int rowoff = (lane < 16) ? 0 : 8;
    const int ncol = lane & 15;
#pragma unroll
    for (int nt = 0; nt < 8; ++nt) {
        const int c = nt * 16 + ncol;
        const float bv = bias[c];
#pragma unroll
        for (int r = 0; r < 8; ++r) {
            long row = mBase + rowoff + r;
            if (row < Mtot) out[row * 128 + c] = acc[nt][r] + bv;
        }
    }
}

// =====================================================================
// Prepack OIHW fp32 weights into bf16 B-fragments.
// Fragment (tap, kc, nt): 32 lanes x 8 dwords; lane l<16: N=l, K=2v,2v+1;
// lane>=16: N=l-16, K=16+2v, 17+2v  (K local to the 32-chunk).
// =====================================================================
__global__ void __launch_bounds__(256)
prepack_w(const float* __restrict__ w, unsigned int* __restrict__ wp,
          int Cin, int ksz)
{
    const int KC = Cin >> 5;
    const long total = (long)ksz * ksz * KC * 8 * 256;
    long idx = (long)blockIdx.x * 256 + threadIdx.x;
    if (idx >= total) return;
    int dw   = (int)(idx & 7);
    int lane = (int)((idx >> 3) & 31);
    int nt   = (int)((idx >> 8) & 7);
    long rest = idx >> 11;                 // tap*KC + kc
    int kc  = (int)(rest % KC);
    int tap = (int)(rest / KC);
    int kh = tap / ksz, kw = tap % ksz;
    int Nloc = lane & 15;
    int K0 = kc * 32 + ((lane >= 16) ? 16 : 0) + 2 * dw;
    int co = nt * 16 + Nloc;
    long w0 = (((long)co * Cin + K0) * ksz + kh) * ksz + kw;
    long w1 = (((long)co * Cin + K0 + 1) * ksz + kh) * ksz + kw;
    unsigned int lo = f2bf(w[w0]);
    unsigned int hi = f2bf(w[w1]);
    wp[idx] = lo | (hi << 16);
}

// =====================================================================
// BN statistics: coalesced, deterministic two-stage reduction.
// C divides 256, so thread t of a 256-thread block always sees channel
// t % C when reading contiguous NHWC elements with a 256*nblk stride.
// =====================================================================
__global__ void __launch_bounds__(256)
bn_stats_partial(const float* __restrict__ y, float* __restrict__ part,
                 long total, int C, int nblk)
{
    const int t = threadIdx.x;
    float s = 0.f, s2 = 0.f;
    const long stride = (long)nblk * 256;
    for (long idx = (long)blockIdx.x * 256 + t; idx < total; idx += stride) {
        float v = y[idx];
        s += v; s2 += v * v;
    }
    __shared__ float ps[256], qs[256];
    ps[t] = s; qs[t] = s2;
    __syncthreads();
    if (t < C) {
        float a = ps[t], b = qs[t];
        if (C == 128) { a += ps[t + 128]; b += qs[t + 128]; }
        part[(long)blockIdx.x * 2 * C + t]     = a;
        part[(long)blockIdx.x * 2 * C + C + t] = b;
    }
}

__global__ void __launch_bounds__(512)
bn_stats_reduce(const float* __restrict__ part, float* __restrict__ stats,
                int C, int nblk)
{
    int c = threadIdx.x;
    if (c >= 2 * C) return;
    float s = 0.f;
    for (int b = 0; b < nblk; ++b) s += part[(long)b * 2 * C + c];
    stats[c] = s;
}

__global__ void
bn_finalize(const float* __restrict__ stats, const float* __restrict__ g,
            const float* __restrict__ be, float* __restrict__ sc,
            float* __restrict__ sh, long Mtot, int C)
{
    int c = threadIdx.x;
    if (c >= C) return;
    float inv = 1.0f / (float)Mtot;
    float mu  = stats[c] * inv;
    float var = stats[C + c] * inv - mu * mu;
    float scale = g[c] * rsqrtf(var + 1e-5f);
    sc[c] = scale;
    sh[c] = be[c] - mu * scale;
}

// flags: 1 = lrelu, 2 = upsample2x, 4 = f32 out (else bf16), 8 = final (sigmoid, NCHW f32)
__global__ void __launch_bounds__(256)
bn_apply(const float* __restrict__ y, const float* __restrict__ sc,
         const float* __restrict__ sh, unsigned short* __restrict__ out_bf,
         float* __restrict__ out_f, long Mtot, int C, int H, int W,
         int Ctot, int choff, int flags)
{
    long idx = (long)blockIdx.x * 256 + threadIdx.x;
    if (idx >= Mtot * (long)C) return;
    int c = (int)(idx % C);
    long m = idx / C;
    float v = y[idx] * sc[c] + sh[c];
    if (flags & 1) v = (v >= 0.f) ? v : 0.2f * v;
    const int HW = H * W;
    int n = (int)(m / HW); int r = (int)(m % HW);
    int h = r / W, w = r % W;
    if (flags & 8) {  // final: sigmoid + upsample2 + NCHW f32
        v = 1.f / (1.f + __expf(-v));
        int Wo = 2 * W;
        long base = (((long)n * Ctot + c) * (2 * H) + 2 * h) * Wo + 2 * w;
        out_f[base] = v; out_f[base + 1] = v;
        out_f[base + Wo] = v; out_f[base + Wo + 1] = v;
        return;
    }
    if (flags & 2) {  // upsample 2x, NHWC
        int Wo = 2 * W;
        long b00 = (((long)n * (2 * H) + 2 * h) * Wo + 2 * w) * Ctot + choff + c;
        long b10 = b00 + (long)Wo * Ctot;
        if (flags & 4) {
            out_f[b00] = v; out_f[b00 + Ctot] = v;
            out_f[b10] = v; out_f[b10 + Ctot] = v;
        } else {
            unsigned short hv = f2bf(v);
            out_bf[b00] = hv; out_bf[b00 + Ctot] = hv;
            out_bf[b10] = hv; out_bf[b10 + Ctot] = hv;
        }
    } else {
        long o = m * Ctot + choff + c;
        if (flags & 4) out_f[o] = v; else out_bf[o] = f2bf(v);
    }
}

// =====================================================================
// get_mask: ConstantPad2d(1, value=1) -> 3x3 stride-2 avg (x4 rounds)
// =====================================================================
__global__ void __launch_bounds__(256)
mask_pool(const float* __restrict__ in, float* __restrict__ out,
          int N, int Hin, int Hout)
{
    long idx = (long)blockIdx.x * 256 + threadIdx.x;
    long tot = (long)N * Hout * Hout;
    if (idx >= tot) return;
    int n = (int)(idx / (Hout * Hout));
    int r = (int)(idx % (Hout * Hout));
    int oh = r / Hout, ow = r % Hout;
    float s = 0.f;
    for (int dh = 0; dh < 3; ++dh)
        for (int dw = 0; dw < 3; ++dw) {
            int ih = 2 * oh + dh - 1, iw = 2 * ow + dw - 1;
            float v = (ih < 0 || ih >= Hin || iw < 0 || iw >= Hin)
                      ? 1.f : in[((long)n * Hin + ih) * Hin + iw];
            s += v;
        }
    out[idx] = s * (1.f / 9.f);
}

// =====================================================================
// NLM over (81,128) with the reference's raw-reshape scramble on in/out.
// One block per batch; X cached in LDS.
// =====================================================================
__global__ void __launch_bounds__(256)
nlm_kernel(const unsigned short* __restrict__ d4,  // NHWC bf16 (b,9,9,128)
           const float* __restrict__ mb,           // pooled mask (b,81) pre-threshold
           unsigned short* __restrict__ r4)        // NHWC bf16 out
{
    __shared__ float X[81 * 128];
    __shared__ float Wv[81];
    __shared__ float Mk[81];
    const int b = blockIdx.x;
    const int tid = threadIdx.x;
    for (int q = tid; q < 81 * 128; q += 256) {
        int ch = q / 81, sp = q % 81;   // scrambled view of NCHW feat
        X[q] = bf2f(d4[((long)b * 81 + sp) * 128 + ch]);
    }
    for (int i = tid; i < 81; i += 256)
        Mk[i] = (mb[b * 81 + i] > (6.f / 9.f)) ? 1.f : 0.f;
    __syncthreads();
    for (int i = 0; i < 81; ++i) {
        if (tid < 81) {
            float d2 = 0.f;
            for (int c = 0; c < 128; ++c) {
                float dd = X[i * 128 + c] - X[tid * 128 + c];
                d2 += dd * dd;
            }
            Wv[tid] = __expf(-d2 * (1.f / 25.f)) * Mk[tid];
        }
        __syncthreads();
        if (tid < 128) {
            float val;
            if (Mk[i] != 0.f) {
                val = X[i * 128 + tid];
            } else {
                float num = 0.f, den = 0.f;
                for (int j = 0; j < 81; ++j) {
                    float wj = Wv[j];
                    num += wj * X[j * 128 + tid];
                    den += wj;
                }
                val = num / den;
            }
            int p = i * 128 + tid;                 // out_flat index
            r4[((long)b * 81 + (p % 81)) * 128 + (p / 81)] = f2bf(val);
        }
        __syncthreads();
    }
}

// NCHW f32 -> NHWC bf16
__global__ void __launch_bounds__(256)
nchw2nhwc_bf16(const float* __restrict__ in, unsigned short* __restrict__ out,
               long total, int C, int HW)
{
    long idx = (long)blockIdx.x * 256 + threadIdx.x;
    if (idx >= total) return;
    int p = (int)(idx % HW);
    long t = idx / HW;
    int c = (int)(t % C);
    int n = (int)(t / C);
    out[((long)n * HW + p) * C + c] = f2bf(in[idx]);
}

// =====================================================================
// Host orchestration
// =====================================================================
extern "C" void kernel_launch(void* const* d_in, const int* in_sizes, int n_in,
                              void* d_out, int out_size, void* d_ws, size_t ws_size,
                              hipStream_t stream)
{
    if (n_in < 108) return;
    auto F = [&](int i) { return (const float*)d_in[i]; };

    struct BP { const float *w1,*b1,*g1,*be1,*w2,*b2,*g2,*be2,*g0,*be0; };
    struct SP { const float *w1,*b1,*g1,*be1; };
    BP d[5], up[5]; SP sk[4];

    const bool schemeA = (in_sizes[2] == 147456);  // insertion-order vs sorted pytree
    const int base = 2;
    if (schemeA) {
        for (int i = 0; i < 5; ++i) { int o = base + i * 8;
            d[i] = {F(o+0),F(o+1),F(o+2),F(o+3),F(o+4),F(o+5),F(o+6),F(o+7),nullptr,nullptr}; }
        for (int i = 0; i < 5; ++i) { int o = base + 40 + i * 10;
            up[i] = {F(o+0),F(o+1),F(o+2),F(o+3),F(o+4),F(o+5),F(o+6),F(o+7),F(o+8),F(o+9)}; }
        for (int i = 0; i < 4; ++i) { int o = base + 90 + i * 4;
            sk[i] = {F(o+0),F(o+1),F(o+2),F(o+3)}; }
    } else {  // sorted pytree: d1..d5, s2..s5, up1..up5; keys sorted within block
        for (int i = 0; i < 5; ++i) { int o = base + i * 8;
            d[i].b1=F(o+0); d[i].b2=F(o+1); d[i].be1=F(o+2); d[i].be2=F(o+3);
            d[i].g1=F(o+4); d[i].g2=F(o+5); d[i].w1=F(o+6); d[i].w2=F(o+7);
            d[i].g0=d[i].be0=nullptr; }
        for (int i = 0; i < 4; ++i) { int o = base + 40 + i * 4;
            sk[i].b1=F(o+0); sk[i].be1=F(o+1); sk[i].g1=F(o+2); sk[i].w1=F(o+3); }
        for (int i = 0; i < 5; ++i) { int o = base + 56 + i * 10;
            up[i].b1=F(o+0); up[i].b2=F(o+1); up[i].be0=F(o+2); up[i].be1=F(o+3);
            up[i].be2=F(o+4); up[i].g0=F(o+5); up[i].g1=F(o+6); up[i].g2=F(o+7);
            up[i].w1=F(o+8); up[i].w2=F(o+9); }
    }

    // ---- workspace bump allocator ----
    char* ws = (char*)d_ws;
    size_t cur = 0;
    auto alloc = [&](size_t bytes) -> char* {
        cur = (cur + 255) & ~(size_t)255;
        char* p = ws + cur; cur += bytes; return p;
    };
    auto wpAlloc = [&](int cin, int k) {
        return (unsigned int*)alloc((size_t)k * k * (cin / 32) * 2048 * 4);
    };

    unsigned int *wpd[5][2], *wpu[5][2], *wps[4];
    for (int i = 0; i < 4; ++i) { wpd[i][0] = wpAlloc(128,3); wpd[i][1] = wpAlloc(128,3); }
    wpd[4][0] = wpAlloc(128,1); wpd[4][1] = wpAlloc(128,1);
    wpu[0][0] = wpAlloc(128,1); wpu[0][1] = wpAlloc(128,1);
    for (int i = 1; i < 5; ++i) { wpu[i][0] = wpAlloc(256,3); wpu[i][1] = wpAlloc(128,3); }
    for (int i = 0; i < 4; ++i) wps[i] = wpAlloc(128,1);

    const long E144 = 8L*144*144*128;
    const int  NBLK = 256;                      // stats partial blocks (fixed -> deterministic)
    unsigned short* X0 = (unsigned short*)alloc((size_t)E144 * 2);
    unsigned short* B1 = (unsigned short*)alloc((size_t)10616832 * 2);
    unsigned short* B2 = (unsigned short*)alloc((size_t)5308416 * 2);
    unsigned short* R1 = (unsigned short*)alloc((size_t)5308416 * 2);
    unsigned short* R2 = (unsigned short*)alloc((size_t)1327104 * 2);
    unsigned short* R3 = (unsigned short*)alloc((size_t)331776 * 2);
    unsigned short* R4 = (unsigned short*)alloc((size_t)82944 * 2);
    unsigned short* D4 = (unsigned short*)alloc((size_t)82944 * 2);
    float* TMPF  = (float*)alloc((size_t)5308416 * 4);
    float* CAT   = (float*)alloc((size_t)10616832 * 4);
    float* MB1   = (float*)alloc((size_t)8*72*72 * 4);
    float* MB2   = (float*)alloc((size_t)8*36*36 * 4);
    float* PART  = (float*)alloc((size_t)NBLK * 512 * 4);
    float* stats = (float*)alloc(512 * 4);
    float* scb   = (float*)alloc(256 * 4);
    float* shb   = (float*)alloc(256 * 4);
    (void)ws_size; (void)out_size;

    // ---- launch helpers ----
    auto prepack = [&](const float* w, unsigned int* wp, int cin, int k) {
        long tot = (long)k * k * (cin / 32) * 2048;
        prepack_w<<<(unsigned)((tot + 255) / 256), 256, 0, stream>>>(w, wp, cin, k);
    };
    auto conv = [&](const unsigned short* in, int Hin, int Cin, int stride, int k,
                    const unsigned int* wp, const float* bias, int Hout) {
        long Mtot = 8L * Hout * Hout;
        conv_wmma<<<(unsigned)((Mtot + 127) / 128), 256, 0, stream>>>(
            in, wp, bias, TMPF, 8, Hin, Hin, Cin, Hout, Hout, stride, k);
    };
    auto bnstat = [&](const float* y, long Mtot, int C, const float* g, const float* be) {
        bn_stats_partial<<<NBLK, 256, 0, stream>>>(y, PART, Mtot * C, C, NBLK);
        bn_stats_reduce<<<1, 512, 0, stream>>>(PART, stats, C, NBLK);
        bn_finalize<<<1, 256, 0, stream>>>(stats, g, be, scb, shb, Mtot, C);
    };
    auto apply = [&](const float* y, long Mtot, int C, int H, int Ctot, int choff,
                     int flags, void* out) {
        long tot = Mtot * C;
        bn_apply<<<(unsigned)((tot + 255) / 256), 256, 0, stream>>>(
            y, scb, shb, (unsigned short*)out, (float*)out,
            Mtot, C, H, H, Ctot, choff, flags);
    };

    // ---- weight prepacking (every call; deterministic) ----
    for (int i = 0; i < 5; ++i) {
        int k = (i < 4) ? 3 : 1;
        prepack(d[i].w1, wpd[i][0], 128, k);
        prepack(d[i].w2, wpd[i][1], 128, k);
    }
    prepack(up[0].w1, wpu[0][0], 128, 1);
    prepack(up[0].w2, wpu[0][1], 128, 1);
    for (int i = 1; i < 5; ++i) {
        prepack(up[i].w1, wpu[i][0], 256, 3);
        prepack(up[i].w2, wpu[i][1], 128, 3);
    }
    for (int i = 0; i < 4; ++i) prepack(sk[i].w1, wps[i], 128, 1);

    // ---- input conversion & mask pyramid ----
    nchw2nhwc_bf16<<<(unsigned)((E144 + 255) / 256), 256, 0, stream>>>(
        (const float*)d_in[1], X0, E144, 128, 144 * 144);
    mask_pool<<<(unsigned)((8L*72*72 + 255)/256), 256, 0, stream>>>((const float*)d_in[0], MB1, 8, 144, 72);
    mask_pool<<<(unsigned)((8L*36*36 + 255)/256), 256, 0, stream>>>(MB1, MB2, 8, 72, 36);
    mask_pool<<<(unsigned)((8L*18*18 + 255)/256), 256, 0, stream>>>(MB2, MB1, 8, 36, 18);
    mask_pool<<<(unsigned)((8L*9*9   + 255)/256), 256, 0, stream>>>(MB1, MB2, 8, 18, 9);

    const long M72 = 8L*72*72, M36 = 8L*36*36, M18 = 8L*18*18, M9 = 8L*81;

    // ---- encoder ----
    conv(X0, 144, 128, 2, 3, wpd[0][0], d[0].b1, 72);
    bnstat(TMPF, M72, 128, d[0].g1, d[0].be1); apply(TMPF, M72, 128, 72, 128, 0, 1, B1);
    conv(B1, 72, 128, 1, 3, wpd[0][1], d[0].b2, 72);
    bnstat(TMPF, M72, 128, d[0].g2, d[0].be2); apply(TMPF, M72, 128, 72, 128, 0, 1, R1);

    conv(R1, 72, 128, 2, 3, wpd[1][0], d[1].b1, 36);
    bnstat(TMPF, M36, 128, d[1].g1, d[1].be1); apply(TMPF, M36, 128, 36, 128, 0, 1, B1);
    conv(B1, 36, 128, 1, 3, wpd[1][1], d[1].b2, 36);
    bnstat(TMPF, M36, 128, d[1].g2, d[1].be2); apply(TMPF, M36, 128, 36, 128, 0, 1, R2);

    conv(R2, 36, 128, 2, 3, wpd[2][0], d[2].b1, 18);
    bnstat(TMPF, M18, 128, d[2].g1, d[2].be1); apply(TMPF, M18, 128, 18, 128, 0, 1, B1);
    conv(B1, 18, 128, 1, 3, wpd[2][1], d[2].b2, 18);
    bnstat(TMPF, M18, 128, d[2].g2, d[2].be2); apply(TMPF, M18, 128, 18, 128, 0, 1, R3);

    conv(R3, 18, 128, 2, 3, wpd[3][0], d[3].b1, 9);
    bnstat(TMPF, M9, 128, d[3].g1, d[3].be1); apply(TMPF, M9, 128, 9, 128, 0, 1, B1);
    conv(B1, 9, 128, 1, 3, wpd[3][1], d[3].b2, 9);
    bnstat(TMPF, M9, 128, d[3].g2, d[3].be2); apply(TMPF, M9, 128, 9, 128, 0, 1, D4);

    // ---- NLM ----
    nlm_kernel<<<8, 256, 0, stream>>>(D4, MB2, R4);

    // ---- d5 (1x1) ----
    conv(R4, 9, 128, 1, 1, wpd[4][0], d[4].b1, 9);
    bnstat(TMPF, M9, 128, d[4].g1, d[4].be1); apply(TMPF, M9, 128, 9, 128, 0, 1, B1);
    conv(B1, 9, 128, 1, 1, wpd[4][1], d[4].b2, 9);
    bnstat(TMPF, M9, 128, d[4].g2, d[4].be2); apply(TMPF, M9, 128, 9, 128, 0, 1|4, CAT);

    // ---- up1 (1x1, no upsample) ----
    bnstat(CAT, M9, 128, up[0].g0, up[0].be0); apply(CAT, M9, 128, 9, 128, 0, 0, B1);
    conv(B1, 9, 128, 1, 1, wpu[0][0], up[0].b1, 9);
    bnstat(TMPF, M9, 128, up[0].g1, up[0].be1); apply(TMPF, M9, 128, 9, 128, 0, 1, B2);
    conv(B2, 9, 128, 1, 1, wpu[0][1], up[0].b2, 9);
    bnstat(TMPF, M9, 128, up[0].g2, up[0].be2); apply(TMPF, M9, 128, 9, 256, 0, 1|4, CAT);
    // s2
    conv(R4, 9, 128, 1, 1, wps[0], sk[0].b1, 9);
    bnstat(TMPF, M9, 128, sk[0].g1, sk[0].be1); apply(TMPF, M9, 128, 9, 256, 128, 1|4, CAT);

    // ---- up2 -> 18 ----
    bnstat(CAT, M9, 256, up[1].g0, up[1].be0); apply(CAT, M9, 256, 9, 256, 0, 0, B1);
    conv(B1, 9, 256, 1, 3, wpu[1][0], up[1].b1, 9);
    bnstat(TMPF, M9, 128, up[1].g1, up[1].be1); apply(TMPF, M9, 128, 9, 128, 0, 1, B2);
    conv(B2, 9, 128, 1, 3, wpu[1][1], up[1].b2, 9);
    bnstat(TMPF, M9, 128, up[1].g2, up[1].be2); apply(TMPF, M9, 128, 9, 256, 0, 1|2|4, CAT);
    // s3
    conv(R3, 18, 128, 1, 1, wps[1], sk[1].b1, 18);
    bnstat(TMPF, M18, 128, sk[1].g1, sk[1].be1); apply(TMPF, M18, 128, 18, 256, 128, 1|4, CAT);

    // ---- up3 -> 36 ----
    bnstat(CAT, M18, 256, up[2].g0, up[2].be0); apply(CAT, M18, 256, 18, 256, 0, 0, B1);
    conv(B1, 18, 256, 1, 3, wpu[2][0], up[2].b1, 18);
    bnstat(TMPF, M18, 128, up[2].g1, up[2].be1); apply(TMPF, M18, 128, 18, 128, 0, 1, B2);
    conv(B2, 18, 128, 1, 3, wpu[2][1], up[2].b2, 18);
    bnstat(TMPF, M18, 128, up[2].g2, up[2].be2); apply(TMPF, M18, 128, 18, 256, 0, 1|2|4, CAT);
    // s4
    conv(R2, 36, 128, 1, 1, wps[2], sk[2].b1, 36);
    bnstat(TMPF, M36, 128, sk[2].g1, sk[2].be1); apply(TMPF, M36, 128, 36, 256, 128, 1|4, CAT);

    // ---- up4 -> 72 ----
    bnstat(CAT, M36, 256, up[3].g0, up[3].be0); apply(CAT, M36, 256, 36, 256, 0, 0, B1);
    conv(B1, 36, 256, 1, 3, wpu[3][0], up[3].b1, 36);
    bnstat(TMPF, M36, 128, up[3].g1, up[3].be1); apply(TMPF, M36, 128, 36, 128, 0, 1, B2);
    conv(B2, 36, 128, 1, 3, wpu[3][1], up[3].b2, 36);
    bnstat(TMPF, M36, 128, up[3].g2, up[3].be2); apply(TMPF, M36, 128, 36, 256, 0, 1|2|4, CAT);
    // s5
    conv(R1, 72, 128, 1, 1, wps[3], sk[3].b1, 72);
    bnstat(TMPF, M72, 128, sk[3].g1, sk[3].be1); apply(TMPF, M72, 128, 72, 256, 128, 1|4, CAT);

    // ---- up5 -> final (BN + lrelu + upsample2 + sigmoid, NCHW f32) ----
    bnstat(CAT, M72, 256, up[4].g0, up[4].be0); apply(CAT, M72, 256, 72, 256, 0, 0, B1);
    conv(B1, 72, 256, 1, 3, wpu[4][0], up[4].b1, 72);
    bnstat(TMPF, M72, 128, up[4].g1, up[4].be1); apply(TMPF, M72, 128, 72, 128, 0, 1, B2);
    conv(B2, 72, 128, 1, 3, wpu[4][1], up[4].b2, 72);
    bnstat(TMPF, M72, 128, up[4].g2, up[4].be2);
    apply(TMPF, M72, 128, 72, 128, 0, 1|8, d_out);
}
